// Attention_3710851743995
// MI455X (gfx1250) — matrix operand order repaired and checked by reading, not compiled, not run on hardware
//
#include <hip/hip_runtime.h>
#include <math.h>

#define D_MODEL 1024
#define NHEAD   16
#define HDIM    64
#define DFF_    4096
#define SEQ     2048
#define BATCH   2
#define LN_EPS  1e-5f

typedef __bf16 bf16;
typedef __attribute__((ext_vector_type(16))) __bf16 v16bf;
typedef __attribute__((ext_vector_type(8)))  __bf16 v8bf;
typedef __attribute__((ext_vector_type(8)))  float  v8f;

union BF16Frag { v16bf v; v8bf h[2]; };

__device__ __forceinline__ v8f wmma_bf16(v16bf a, v16bf b, v8f c) {
  return __builtin_amdgcn_wmma_f32_16x16x32_bf16(false, a, false, b, (short)0, c,
                                                 false, false);
}

// Async global->LDS copy of 16 bytes per lane (gfx1250 ASYNCcnt path).
// Generic LDS pointers on CDNA5 carry the LDS offset in addr[31:0].
__device__ __forceinline__ void async_ld16(void* lds, const void* g) {
  asm volatile("global_load_async_to_lds_b128 %0, %1, off"
               :
               : "v"((unsigned)(size_t)lds), "v"((unsigned long long)(size_t)g)
               : "memory");
}
__device__ __forceinline__ void async_wait0() {
  asm volatile("s_wait_asynccnt 0x0" ::: "memory");
}

// ---------------------------------------------------------------------------
// fp32 -> bf16 straight conversion (8 elements / thread).
// ---------------------------------------------------------------------------
__global__ __launch_bounds__(256)
void cvt_kernel(const float* __restrict__ src, bf16* __restrict__ dst) {
  const size_t base = ((size_t)blockIdx.x * 256 + threadIdx.x) * 8;
  const float4 f0 = *(const float4*)(src + base);
  const float4 f1 = *(const float4*)(src + base + 4);
  v8bf o;
  o[0] = (bf16)f0.x; o[1] = (bf16)f0.y; o[2] = (bf16)f0.z; o[3] = (bf16)f0.w;
  o[4] = (bf16)f1.x; o[5] = (bf16)f1.y; o[6] = (bf16)f1.z; o[7] = (bf16)f1.w;
  *(v8bf*)(dst + base) = o;
}

// ---------------------------------------------------------------------------
// Weight transpose+convert: W[K,N] fp32 -> Wt[N,K] bf16 (64x64 tiles).
// Makes GEMM B-tile staging a contiguous-k straight copy.
// ---------------------------------------------------------------------------
__global__ __launch_bounds__(256)
void transpose_cvt_kernel(const float* __restrict__ W, bf16* __restrict__ Wt,
                          int K, int N) {
  __shared__ alignas(16) bf16 T[64 * 72];
  const int t  = threadIdx.x;
  const int n0 = blockIdx.x * 64;
  const int k0 = blockIdx.y * 64;
  {
    const int kr  = t >> 2;
    const int nc0 = (t & 3) * 16;
    const float4* src = (const float4*)(W + (size_t)(k0 + kr) * N + n0 + nc0);
#pragma unroll
    for (int u = 0; u < 4; ++u) {
      float4 f = src[u];
      T[(nc0 + 4 * u + 0) * 72 + kr] = (bf16)f.x;
      T[(nc0 + 4 * u + 1) * 72 + kr] = (bf16)f.y;
      T[(nc0 + 4 * u + 2) * 72 + kr] = (bf16)f.z;
      T[(nc0 + 4 * u + 3) * 72 + kr] = (bf16)f.w;
    }
  }
  __syncthreads();
  {
    const int n   = t >> 2;
    const int kc0 = (t & 3) * 16;
    bf16* dst = Wt + (size_t)(n0 + n) * K + k0 + kc0;
    *(v8bf*)dst       = *(const v8bf*)(T + n * 72 + kc0);
    *(v8bf*)(dst + 8) = *(const v8bf*)(T + n * 72 + kc0 + 8);
  }
}

// ---------------------------------------------------------------------------
// V relayout: Vb[tok, D] bf16 -> Vt[b, h, hd, seq] bf16 (per-head transpose),
// so attention V tiles become contiguous-key straight copies.
// ---------------------------------------------------------------------------
__global__ __launch_bounds__(256)
void vtrans_kernel(const bf16* __restrict__ Vb, bf16* __restrict__ Vt) {
  __shared__ alignas(16) bf16 T[64 * 72];
  const int t  = threadIdx.x;
  const int kt = blockIdx.x * 64;
  const int h  = blockIdx.y;
  const int b  = blockIdx.z;
  const size_t tokBase = (size_t)b * SEQ;
  {
    const int key = t >> 2;
    const int hd0 = (t & 3) * 16;
    const bf16* src =
        Vb + (tokBase + kt + key) * D_MODEL + h * HDIM + hd0;
    const v8bf a = *(const v8bf*)src;
    const v8bf c = *(const v8bf*)(src + 8);
#pragma unroll
    for (int j = 0; j < 8; ++j) {
      T[(hd0 + j) * 72 + key]     = a[j];
      T[(hd0 + 8 + j) * 72 + key] = c[j];
    }
  }
  __syncthreads();
  {
    const int hd   = t >> 2;
    const int koff = (t & 3) * 16;
    bf16* dst = Vt + (((size_t)b * NHEAD + h) * HDIM + hd) * SEQ + kt + koff;
    *(v8bf*)dst       = *(const v8bf*)(T + hd * 72 + koff);
    *(v8bf*)(dst + 8) = *(const v8bf*)(T + hd * 72 + koff + 8);
  }
}

// ---------------------------------------------------------------------------
// Tiled GEMM: act(A[M,K]bf16 @ Wt[N,K]bf16^T + bias) -> Cf fp32 and/or
// Cb bf16*oscale. Tile 128x128x32, 8 waves. Double-buffered async-to-LDS
// staging: tile k+1 DMA overlaps tile k WMMAs. act: 0 = none, 1 = exact GELU.
// ---------------------------------------------------------------------------
__global__ __launch_bounds__(256)
void gemm_bf16_kernel(const bf16* __restrict__ A, const bf16* __restrict__ Wt,
                      const float* __restrict__ bias, float* __restrict__ Cf,
                      bf16* __restrict__ Cb, int M, int N, int K, int act,
                      float oscale) {
  __shared__ alignas(32) bf16 As[2][128 * 32];   // [buf][row][k]
  __shared__ alignas(32) bf16 Bs[2][128 * 32];   // [buf][n][k]

  const int tid   = threadIdx.x;
  const int lane  = tid & 31;
  const int wave  = tid >> 5;
  const int wm    = wave & 3;
  const int wn    = wave >> 2;
  const int lhalf = lane >> 4;
  const int l15   = lane & 15;
  const int rowBase = blockIdx.y * 128;
  const int colBase = blockIdx.x * 128;

  v8f acc[2][4];
#pragma unroll
  for (int i = 0; i < 2; ++i)
#pragma unroll
    for (int j = 0; j < 4; ++j)
#pragma unroll
      for (int e = 0; e < 8; ++e) acc[i][j][e] = 0.0f;

  auto stage = [&](int buf, int k0) {
#pragma unroll
    for (int u = 0; u < 2; ++u) {
      const int c = tid + 256 * u;
      const int r = c >> 2, ko = (c & 3) * 8;
      async_ld16(&As[buf][r * 32 + ko],
                 A + (size_t)(rowBase + r) * K + k0 + ko);
    }
#pragma unroll
    for (int u = 0; u < 2; ++u) {
      const int c = tid + 256 * u;
      const int n = c >> 2, ko = (c & 3) * 8;
      async_ld16(&Bs[buf][n * 32 + ko],
                 Wt + (size_t)(colBase + n) * K + k0 + ko);
    }
  };

  stage(0, 0);
  int cur = 0;
  for (int k0 = 0; k0 < K; k0 += 32) {
    async_wait0();        // current buffer's DMA complete (this wave)
    __syncthreads();      // ... and every wave's; prior reads of other buf done
    if (k0 + 32 < K) stage(cur ^ 1, k0 + 32);  // overlap next DMA with compute

    BF16Frag aF[2];
#pragma unroll
    for (int i = 0; i < 2; ++i) {
      const bf16* p = &As[cur][(32 * wm + 16 * i + l15) * 32 + 16 * lhalf];
      aF[i].h[0] = *(const v8bf*)p;
      aF[i].h[1] = *(const v8bf*)(p + 8);
    }
#pragma unroll
    for (int j = 0; j < 4; ++j) {
      const v16bf bF = *(const v16bf*)&Bs[cur][(64 * wn + 16 * j + l15) * 32 +
                                              16 * lhalf];
#pragma unroll
      for (int i = 0; i < 2; ++i) acc[i][j] = wmma_bf16(aF[i].v, bF, acc[i][j]);
    }
    cur ^= 1;
  }

#pragma unroll
  for (int i = 0; i < 2; ++i)
#pragma unroll
    for (int j = 0; j < 4; ++j)
#pragma unroll
      for (int e = 0; e < 8; ++e) {
        const int row = rowBase + 32 * wm + 16 * i + 8 * lhalf + e;
        const int col = colBase + 64 * wn + 16 * j + l15;
        float v = acc[i][j][e] + bias[col];
        if (act == 1) v = 0.5f * v * (1.0f + erff(v * 0.70710678118654752f));
        const size_t idx = (size_t)row * N + col;
        if (Cf) Cf[idx] = v;
        if (Cb) Cb[idx] = (bf16)(v * oscale);
      }
}

// ---------------------------------------------------------------------------
// Flash attention: 128 threads = 4 waves, 64 query rows per block. K/V tiles
// double-buffered via async-to-LDS; online softmax; WMMA for Q·K^T and P·V.
// Q is pre-scaled by 1/sqrt(HD) at projection time.
// ---------------------------------------------------------------------------
__global__ __launch_bounds__(128)
void attn_kernel(const bf16* __restrict__ Q, const bf16* __restrict__ Kx,
                 const bf16* __restrict__ Vt, float* __restrict__ O) {
  __shared__ alignas(32) bf16 Ks[2][64 * 64];   // [buf][key][hd]
  __shared__ alignas(32) bf16 Vs[2][64 * 64];   // [buf][hd][key]
  __shared__ alignas(32) bf16 Ps[4 * 16 * 64];  // per-wave [row][key]

  const int tid   = threadIdx.x;
  const int lane  = tid & 31;
  const int wave  = tid >> 5;
  const int lhalf = lane >> 4;
  const int l15   = lane & 15;
  const int h     = blockIdx.y;
  const int b     = blockIdx.z;
  const size_t tokBase = (size_t)b * SEQ;
  const int hOff  = h * HDIM;
  const size_t vhBase = ((size_t)b * NHEAD + h) * HDIM;  // row base into Vt
  const int qRow0 = blockIdx.x * 64 + wave * 16;

  BF16Frag qF[2];
#pragma unroll
  for (int kc = 0; kc < 2; ++kc) {
    const bf16* p =
        Q + (tokBase + qRow0 + l15) * D_MODEL + hOff + kc * 32 + 16 * lhalf;
    qF[kc].h[0] = *(const v8bf*)p;
    qF[kc].h[1] = *(const v8bf*)(p + 8);
  }

  float mrow[8], lrow[8];
  v8f oF[4];
#pragma unroll
  for (int e = 0; e < 8; ++e) { mrow[e] = -1e30f; lrow[e] = 0.0f; }
#pragma unroll
  for (int nc = 0; nc < 4; ++nc)
#pragma unroll
    for (int e = 0; e < 8; ++e) oF[nc][e] = 0.0f;

  bf16* PsW = Ps + wave * 16 * 64;

  auto stageKV = [&](int buf, int kt) {
#pragma unroll
    for (int u = 0; u < 4; ++u) {
      const int c = tid + 128 * u;
      const int key = c >> 3, ko = (c & 7) * 8;
      async_ld16(&Ks[buf][key * 64 + ko],
                 Kx + (tokBase + kt + key) * D_MODEL + hOff + ko);
    }
#pragma unroll
    for (int u = 0; u < 4; ++u) {
      const int c = tid + 128 * u;
      const int hd = c >> 3, ko = (c & 7) * 8;
      async_ld16(&Vs[buf][hd * 64 + ko], Vt + (vhBase + hd) * SEQ + kt + ko);
    }
  };

  stageKV(0, 0);
  int cur = 0;
  for (int kt = 0; kt < SEQ; kt += 64) {
    async_wait0();
    __syncthreads();
    if (kt + 64 < SEQ) stageKV(cur ^ 1, kt + 64);  // overlap DMA with compute

    // scores S = Q K^T (16 x 64 per wave)
    v8f sF[4];
#pragma unroll
    for (int nj = 0; nj < 4; ++nj) {
      v8f s;
#pragma unroll
      for (int e = 0; e < 8; ++e) s[e] = 0.0f;
#pragma unroll
      for (int kc = 0; kc < 2; ++kc) {
        const v16bf bK = *(const v16bf*)&Ks[cur][(16 * nj + l15) * 64 +
                                                 kc * 32 + 16 * lhalf];
        s = wmma_bf16(qF[kc].v, bK, s);
      }
      sF[nj] = s;
    }

    // online softmax (row spans 16 lanes of a half + 4 frags)
    float fac[8];
#pragma unroll
    for (int e = 0; e < 8; ++e) {
      float v = fmaxf(fmaxf(sF[0][e], sF[1][e]), fmaxf(sF[2][e], sF[3][e]));
      v = fmaxf(v, __shfl_xor(v, 1, 32));
      v = fmaxf(v, __shfl_xor(v, 2, 32));
      v = fmaxf(v, __shfl_xor(v, 4, 32));
      v = fmaxf(v, __shfl_xor(v, 8, 32));
      const float mnew = fmaxf(mrow[e], v);
      fac[e] = expf(mrow[e] - mnew);
      mrow[e] = mnew;
      lrow[e] *= fac[e];
    }
#pragma unroll
    for (int nc = 0; nc < 4; ++nc)
#pragma unroll
      for (int e = 0; e < 8; ++e) oF[nc][e] *= fac[e];

    float tsum[8];
#pragma unroll
    for (int e = 0; e < 8; ++e) tsum[e] = 0.0f;
#pragma unroll
    for (int nj = 0; nj < 4; ++nj)
#pragma unroll
      for (int e = 0; e < 8; ++e) {
        const float p = expf(sF[nj][e] - mrow[e]);
        tsum[e] += p;
        PsW[(8 * lhalf + e) * 64 + 16 * nj + l15] = (bf16)p;
      }
#pragma unroll
    for (int e = 0; e < 8; ++e) {
      float t = tsum[e];
      t += __shfl_xor(t, 1, 32);
      t += __shfl_xor(t, 2, 32);
      t += __shfl_xor(t, 4, 32);
      t += __shfl_xor(t, 8, 32);
      lrow[e] += t;
    }

    // O += P @ V
#pragma unroll
    for (int kc = 0; kc < 2; ++kc) {
      BF16Frag aP;
      const bf16* p = PsW + l15 * 64 + kc * 32 + 16 * lhalf;
      aP.h[0] = *(const v8bf*)p;
      aP.h[1] = *(const v8bf*)(p + 8);
#pragma unroll
      for (int nc = 0; nc < 4; ++nc) {
        const v16bf bV = *(const v16bf*)&Vs[cur][(16 * nc + l15) * 64 +
                                                 kc * 32 + 16 * lhalf];
        oF[nc] = wmma_bf16(aP.v, bV, oF[nc]);
      }
    }
    cur ^= 1;
  }

#pragma unroll
  for (int nc = 0; nc < 4; ++nc)
#pragma unroll
    for (int e = 0; e < 8; ++e) {
      const int row = qRow0 + 8 * lhalf + e;
      const int col = hOff + 16 * nc + l15;
      O[(tokBase + row) * D_MODEL + col] = oF[nc][e] / lrow[e];
    }
}

// ---------------------------------------------------------------------------
// outf = LayerNorm(a + b)*g + be  (fp32), optional bf16 copy outb.
// ---------------------------------------------------------------------------
__global__ __launch_bounds__(256)
void add_ln_kernel(const float* __restrict__ a, const float* __restrict__ b,
                   const float* __restrict__ g, const float* __restrict__ be,
                   float* __restrict__ outf, bf16* __restrict__ outb) {
  __shared__ float red[256];
  const size_t row = blockIdx.x;
  const int tid = threadIdx.x;

  float v[4];
  float s = 0.0f, sq = 0.0f;
#pragma unroll
  for (int u = 0; u < 4; ++u) {
    const int c = tid + 256 * u;
    const float x = a[row * D_MODEL + c] + b[row * D_MODEL + c];
    v[u] = x; s += x; sq += x * x;
  }
  red[tid] = s;
  __syncthreads();
  for (int off = 128; off > 0; off >>= 1) {
    if (tid < off) red[tid] += red[tid + off];
    __syncthreads();
  }
  const float mu = red[0] * (1.0f / D_MODEL);
  __syncthreads();
  red[tid] = sq;
  __syncthreads();
  for (int off = 128; off > 0; off >>= 1) {
    if (tid < off) red[tid] += red[tid + off];
    __syncthreads();
  }
  const float var = red[0] * (1.0f / D_MODEL) - mu * mu;
  const float rstd = rsqrtf(var + LN_EPS);
#pragma unroll
  for (int u = 0; u < 4; ++u) {
    const int c = tid + 256 * u;
    const float y = (v[u] - mu) * rstd * g[c] + be[c];
    outf[row * D_MODEL + c] = y;
    if (outb) outb[row * D_MODEL + c] = (bf16)y;
  }
}

// ---------------------------------------------------------------------------
extern "C" void kernel_launch(void* const* d_in, const int* in_sizes, int n_in,
                              void* d_out, int out_size, void* d_ws,
                              size_t ws_size, hipStream_t stream) {
  const float* x   = (const float*)d_in[0];
  const float* Wq  = (const float*)d_in[1];
  const float* bq  = (const float*)d_in[2];
  const float* Wk  = (const float*)d_in[3];
  const float* bk  = (const float*)d_in[4];
  const float* Wv  = (const float*)d_in[5];
  const float* bv  = (const float*)d_in[6];
  const float* W1  = (const float*)d_in[7];
  const float* b1  = (const float*)d_in[8];
  const float* W2  = (const float*)d_in[9];
  const float* b2  = (const float*)d_in[10];
  const float* g1  = (const float*)d_in[11];
  const float* be1 = (const float*)d_in[12];
  const float* g2  = (const float*)d_in[13];
  const float* be2 = (const float*)d_in[14];
  float* out = (float*)d_out;

  const int    M   = BATCH * SEQ;             // 4096
  const size_t tok = (size_t)M * D_MODEL;     // 4,194,304 elements

  unsigned char* w = (unsigned char*)d_ws;
  auto alloc = [&](size_t bytes) {
    unsigned char* p = w;
    w += (bytes + 255) & ~(size_t)255;
    return p;
  };
  bf16*  xb  = (bf16*)alloc(tok * 2);
  bf16*  Wqt = (bf16*)alloc((size_t)D_MODEL * D_MODEL * 2);
  bf16*  Wkt = (bf16*)alloc((size_t)D_MODEL * D_MODEL * 2);
  bf16*  Wvt = (bf16*)alloc((size_t)D_MODEL * D_MODEL * 2);
  bf16*  W1t = (bf16*)alloc((size_t)D_MODEL * DFF_ * 2);
  bf16*  W2t = (bf16*)alloc((size_t)D_MODEL * DFF_ * 2);
  bf16*  Qb  = (bf16*)alloc(tok * 2);
  bf16*  Kb  = (bf16*)alloc(tok * 2);
  bf16*  Vb  = (bf16*)alloc(tok * 2);
  bf16*  Vt  = (bf16*)alloc(tok * 2);
  float* AO  = (float*)alloc(tok * 4);
  float* R1f = (float*)alloc(tok * 4);
  bf16*  R1b = (bf16*)alloc(tok * 2);
  bf16*  F1b = (bf16*)alloc((size_t)M * DFF_ * 2);
  float* F2f = (float*)alloc(tok * 4);

  const dim3 blk(256);

  // one-time conversions / relayouts
  cvt_kernel<<<dim3((unsigned)(tok / 2048)), blk, 0, stream>>>(x, xb);
  transpose_cvt_kernel<<<dim3(D_MODEL / 64, D_MODEL / 64), blk, 0, stream>>>(
      Wq, Wqt, D_MODEL, D_MODEL);
  transpose_cvt_kernel<<<dim3(D_MODEL / 64, D_MODEL / 64), blk, 0, stream>>>(
      Wk, Wkt, D_MODEL, D_MODEL);
  transpose_cvt_kernel<<<dim3(D_MODEL / 64, D_MODEL / 64), blk, 0, stream>>>(
      Wv, Wvt, D_MODEL, D_MODEL);
  transpose_cvt_kernel<<<dim3(DFF_ / 64, D_MODEL / 64), blk, 0, stream>>>(
      W1, W1t, D_MODEL, DFF_);
  transpose_cvt_kernel<<<dim3(D_MODEL / 64, DFF_ / 64), blk, 0, stream>>>(
      W2, W2t, DFF_, D_MODEL);

  const dim3 gD(D_MODEL / 128, M / 128);      // (8, 32)
  const dim3 gF(DFF_ / 128, M / 128);         // (32, 32)

  // Q/K/V projections (bf16 out; Q pre-scaled by 1/sqrt(HD))
  gemm_bf16_kernel<<<gD, blk, 0, stream>>>(xb, Wqt, bq, nullptr, Qb, M, D_MODEL,
                                           D_MODEL, 0, 0.125f);
  gemm_bf16_kernel<<<gD, blk, 0, stream>>>(xb, Wkt, bk, nullptr, Kb, M, D_MODEL,
                                           D_MODEL, 0, 1.0f);
  gemm_bf16_kernel<<<gD, blk, 0, stream>>>(xb, Wvt, bv, nullptr, Vb, M, D_MODEL,
                                           D_MODEL, 0, 1.0f);
  vtrans_kernel<<<dim3(SEQ / 64, NHEAD, BATCH), blk, 0, stream>>>(Vb, Vt);

  attn_kernel<<<dim3(SEQ / 64, NHEAD, BATCH), dim3(128), 0, stream>>>(Qb, Kb,
                                                                      Vt, AO);
  // res1 = LN(attn + x): fp32 for residual, bf16 for FFN input
  add_ln_kernel<<<dim3(M), blk, 0, stream>>>(AO, x, g1, be1, R1f, R1b);
  // FFN
  gemm_bf16_kernel<<<gF, blk, 0, stream>>>(R1b, W1t, b1, nullptr, F1b, M, DFF_,
                                           D_MODEL, 1, 1.0f);
  gemm_bf16_kernel<<<gD, blk, 0, stream>>>(F1b, W2t, b2, F2f, nullptr, M,
                                           D_MODEL, DFF_, 0, 1.0f);
  add_ln_kernel<<<dim3(M), blk, 0, stream>>>(F2f, R1f, g2, be2, out, nullptr);
}
